// RNN_18992345383093
// MI455X (gfx1250) — matrix operand order repaired and checked by reading, not compiled, not run on hardware
//
#include <hip/hip_runtime.h>
#include <hip/hip_bf16.h>

typedef __attribute__((ext_vector_type(16))) __bf16 bf16x16;
typedef __attribute__((ext_vector_type(8)))  float  f32x8;

#define DD     512
#define VV     128
#define FF     4
#define NROWS  16384
#define G3     1536
#define LSTR   520            // padded LDS row stride (bf16 elems): 1040B -> 4-bank row shift, conflict-free
#define KBLK   16             // 512 / 32
#define NT_V   8              // 128 / 16
#define WAVES  8
#define PAIRS  4              // wave pairs per WG; each pair owns a 32-row tile
#define H2TILE (32 * LSTR)                       // one 32-row h buffer (bf16 elems)
#define LDS_BYTES (PAIRS * 2 * H2TILE * 2)       // 266240 B, < 320KB WGP limit

// packed weight geometry (bf16 elements)
#define FRAG_EL   512                            // elements per fragment (32 lanes x 16)
#define JT_EL     (16 * FRAG_EL)                 // one ntile row of 16 kblocks = 8192 el
#define GATE_EL   (32 * JT_EL)                   // 32 ntiles (one gate) = 262144 el
#define WHH_EL    (3 * GATE_EL)                  // w_hh base offset = 786432 el

// ---------- bf16 helpers (RNE) ----------
__device__ __forceinline__ unsigned short f2bf(float f) {
  unsigned u = __builtin_bit_cast(unsigned, f);
  unsigned r = (u + 0x7FFFu + ((u >> 16) & 1u)) >> 16;
  return (unsigned short)r;
}
__device__ __forceinline__ float bf2f(unsigned short s) {
  return __builtin_bit_cast(float, ((unsigned)s) << 16);
}
__device__ __forceinline__ float fast_sigmoid(float x) {
  return __builtin_amdgcn_rcpf(1.f + __expf(-x));
}
__device__ __forceinline__ float fast_tanh(float x) {
  float e2 = __expf(-2.f * x);
  return (1.f - e2) * __builtin_amdgcn_rcpf(1.f + e2);
}

// Load one 32B-per-lane packed fragment (two b128s, immediate-offset friendly).
__device__ __forceinline__ bf16x16 ldFrag(const unsigned short* __restrict__ p) {
  union { uint4 q[2]; bf16x16 v; } u;
  u.q[0] = *(const uint4*)p;
  u.q[1] = *(const uint4*)(p + 8);
  return u.v;
}

#define WMMA_BF16(A, B, C) __builtin_amdgcn_wmma_f32_16x16x32_bf16(false, (A), false, (B), (short)0, (C), false, false)

// ---------- pack kernel: fp32 weights -> bf16 WMMA B-fragments ----------
// B layout (16x16x32 bf16, wave32): lane n = l&15, K = 8*(l>>4) + (e<8 ? e : e+8) within a 32-K block.
// Combined buffer: [w_ih gates r|z|n][w_hh gates r|z|n] so the main loop uses one base + imm offsets.
__global__ void pack_frags(const float* __restrict__ w_ih,
                           const float* __restrict__ w_hh,
                           const float* __restrict__ proj_w,
                           unsigned short* __restrict__ wP,     // 2*786432 el
                           unsigned short* __restrict__ projP) {
  int fi = blockIdx.x;      // 0..3583
  int l  = threadIdx.x;     // 0..31
  const float* src;
  unsigned short* dst;
  int kb;
  if (fi < 1536) {                       // w_ih: 96 ntiles x 16 kblocks
    int nt = fi >> 4; kb = fi & 15;
    src = w_ih + (size_t)(nt * 16 + (l & 15)) * DD;
    dst = wP + ((size_t)fi * 32 + l) * 16;
  } else if (fi < 3072) {                // w_hh -> second half of wP
    int f2 = fi - 1536; int nt = f2 >> 4; kb = f2 & 15;
    src = w_hh + (size_t)(nt * 16 + (l & 15)) * DD;
    dst = wP + (size_t)WHH_EL + ((size_t)f2 * 32 + l) * 16;
  } else {                               // proj_w: 4 feats x 8 ntiles x 16 kblocks
    int f3 = fi - 3072; int ff = f3 >> 7; int rem = f3 & 127;
    int nt = rem >> 4; kb = rem & 15;
    src = proj_w + (size_t)(ff * VV + nt * 16 + (l & 15)) * DD;
    dst = projP + ((size_t)f3 * 32 + l) * 16;
  }
  int kh = kb * 32 + 8 * (l >> 4);
  #pragma unroll
  for (int e = 0; e < 16; ++e) dst[e] = f2bf(src[kh + (e < 8 ? e : e + 8)]);
}

// ---------- posG[f][g] = pos_enc[f] . w_ih[g] + b_ih[g] ----------
__global__ void pack_posg(const float* __restrict__ pos_enc,
                          const float* __restrict__ w_ih,
                          const float* __restrict__ b_ih,
                          float* __restrict__ posG) {
  int id = blockIdx.x * blockDim.x + threadIdx.x;
  if (id >= FF * G3) return;
  int f = id / G3, g = id - f * G3;
  const float* p = pos_enc + f * DD;
  const float* w = w_ih + (size_t)g * DD;
  float s = b_ih[g];
  for (int d = 0; d < DD; ++d) s += p[d] * w[d];
  posG[id] = s;
}

// ---------- main GRU + projection kernel ----------
// 8 waves/WG = 4 wave-pairs; each pair owns a 32-row tile (M=32 -> 2 WMMAs per B fragment).
// Within a pair, waves split gate columns (jt) and vocab tiles (vt).
__global__ void __launch_bounds__(256, 1)
gru_main(const float* __restrict__ h0g,
         const int*   __restrict__ target,
         const float* __restrict__ emb,
         const float* __restrict__ b_hh,
         const float* __restrict__ proj_b,
         const unsigned short* __restrict__ wP,
         const unsigned short* __restrict__ projP,
         const float* __restrict__ posG,
         float* __restrict__ out) {
  extern __shared__ unsigned short smem[];
  const int lane  = threadIdx.x & 31;
  const int wave  = threadIdx.x >> 5;
  const int sub   = wave & 1;                    // which half of the pair's work
  const int pair  = wave >> 1;
  const int rbase = (blockIdx.x * PAIRS + pair) * 32;   // 32 rows per pair
  const int hbase = pair * (2 * H2TILE);         // pair's LDS slice (elem offset)

  // stage h0 (fp32 -> bf16): each wave fills its 16 rows; 16B LDS stores
  {
    const int rofs = sub * 16;
    const float* src = h0g + (size_t)(rbase + rofs) * DD;
    for (int idx = lane * 8; idx < 16 * DD; idx += 32 * 8) {
      int r = idx >> 9, c = idx & (DD - 1);
      const float* s = src + (size_t)r * DD + c;
      float4 a0 = *(const float4*)(s);
      float4 a1 = *(const float4*)(s + 4);
      union { unsigned short u[8]; uint4 q; } t;
      t.u[0] = f2bf(a0.x); t.u[1] = f2bf(a0.y); t.u[2] = f2bf(a0.z); t.u[3] = f2bf(a0.w);
      t.u[4] = f2bf(a1.x); t.u[5] = f2bf(a1.y); t.u[6] = f2bf(a1.z); t.u[7] = f2bf(a1.w);
      *(uint4*)(&smem[hbase + (rofs + r) * LSTR + c]) = t.q;
    }
  }
  __syncthreads();

  // per-lane embedding row indices: lane gathers rows (lane&15) [tile0] and 16+(lane&15) [tile1]
  const int er0 = rbase + (lane & 15);
  const int er1 = er0 + 16;
  const int idx00 = target[er0 * FF + 0], idx10 = target[er1 * FF + 0];
  const int idx01 = target[er0 * FF + 1], idx11 = target[er1 * FF + 1];
  const int idx02 = target[er0 * FF + 2], idx12 = target[er1 * FF + 2];

  const int mrow  = lane & 15;
  const int khalf = 8 * (lane >> 4);     // A-fragment K half select
  const int mhi   = 8 * (lane >> 4);     // C/D row-half select
  const int subu  = __builtin_amdgcn_readfirstlane(sub);   // wave-uniform in SGPR

  int curo = 0;
  #pragma unroll
  for (int f = 0; f < FF; ++f) {
    const int nxto = curo ^ H2TILE;
    const float* embf0 = nullptr;
    const float* embf1 = nullptr;
    if (f == 1)      { embf0 = emb + (size_t)idx00 * DD;               embf1 = emb + (size_t)idx10 * DD; }
    else if (f == 2) { embf0 = emb + ((size_t)VV + idx01) * DD;        embf1 = emb + ((size_t)VV + idx11) * DD; }
    else if (f == 3) { embf0 = emb + ((size_t)(2 * VV) + idx02) * DD;  embf1 = emb + ((size_t)(2 * VV) + idx12) * DD; }
    const float* pg = posG + f * G3;

    // ---- gates: this wave handles jt in [sub*16, sub*16+16), both 16-row tiles ----
    for (int jj = 0; jj < 16; ++jj) {
      const int jtu = subu * 16 + jj;                        // SGPR-uniform
      // single base pointer; the 5 other fragments live at immediate byte offsets
      const unsigned short* pB0 = wP + (size_t)jtu * JT_EL + (size_t)lane * 16;

      f32x8 accR0 = {}, accZ0 = {}, axn0 = {}, ahn0 = {};
      f32x8 accR1 = {}, accZ1 = {}, axn1 = {}, ahn1 = {};
      for (int kb = 0; kb < KBLK; ++kb) {
        // A_h fragments from LDS (tile0 rows 0..15, tile1 rows 16..31)
        const int ho0 = hbase + curo + mrow * LSTR + kb * 32 + khalf;
        const int ho1 = ho0 + 16 * LSTR;
        union { uint4 q[2]; bf16x16 v; } ah0, ah1;
        ah0.q[0] = *(const uint4*)(&smem[ho0]);
        ah0.q[1] = *(const uint4*)(&smem[ho0 + 16]);
        ah1.q[0] = *(const uint4*)(&smem[ho1]);
        ah1.q[1] = *(const uint4*)(&smem[ho1 + 16]);

        bf16x16 ax0, ax1;
        if (f == 0) {
          ax0 = ah0.v; ax1 = ah1.v;               // x0 == h0 (pos folded into posG)
        } else {
          const float* s0 = embf0 + kb * 32 + khalf;
          const float* s1 = embf1 + kb * 32 + khalf;
          union { unsigned short u[16]; bf16x16 v; } t0, t1;
          #pragma unroll
          for (int e = 0; e < 8; ++e) { t0.u[e] = f2bf(s0[e]); t0.u[8 + e] = f2bf(s0[16 + e]); }
          #pragma unroll
          for (int e = 0; e < 8; ++e) { t1.u[e] = f2bf(s1[e]); t1.u[8 + e] = f2bf(s1[16 + e]); }
          ax0 = t0.v; ax1 = t1.v;
        }

        const unsigned short* pB = pB0 + (size_t)kb * FRAG_EL;
        bf16x16 bir = ldFrag(pB);
        bf16x16 biz = ldFrag(pB + GATE_EL);
        bf16x16 bin = ldFrag(pB + 2 * GATE_EL);
        bf16x16 bhr = ldFrag(pB + WHH_EL);
        bf16x16 bhz = ldFrag(pB + WHH_EL + GATE_EL);
        bf16x16 bhn = ldFrag(pB + WHH_EL + 2 * GATE_EL);

        // 12 WMMAs per 6 B fragments; same-accumulator ops kept 6 apart
        accR0 = WMMA_BF16(ax0,   bir, accR0);
        accR1 = WMMA_BF16(ax1,   bir, accR1);
        accZ0 = WMMA_BF16(ax0,   biz, accZ0);
        accZ1 = WMMA_BF16(ax1,   biz, accZ1);
        axn0  = WMMA_BF16(ax0,   bin, axn0);
        axn1  = WMMA_BF16(ax1,   bin, axn1);
        accR0 = WMMA_BF16(ah0.v, bhr, accR0);
        accR1 = WMMA_BF16(ah1.v, bhr, accR1);
        accZ0 = WMMA_BF16(ah0.v, bhz, accZ0);
        accZ1 = WMMA_BF16(ah1.v, bhz, accZ1);
        ahn0  = WMMA_BF16(ah0.v, bhn, ahn0);
        ahn1  = WMMA_BF16(ah1.v, bhn, ahn1);
      }

      // fused GRU gate epilogue; C/D layout: elem i -> (M = i + 8*(lane>>4), N = lane&15)
      const int col = jtu * 16 + mrow;
      const float pr = pg[col]        + b_hh[col];
      const float pz = pg[512 + col]  + b_hh[512 + col];
      const float pn = pg[1024 + col];
      const float bn = b_hh[1024 + col];
      #pragma unroll
      for (int i = 0; i < 8; ++i) {
        int m0 = i + mhi;
        int m1 = m0 + 16;
        float r0  = fast_sigmoid(accR0[i] + pr);
        float z0  = fast_sigmoid(accZ0[i] + pz);
        float nn0 = fast_tanh(axn0[i] + pn + r0 * (ahn0[i] + bn));
        float ho0 = bf2f(smem[hbase + curo + m0 * LSTR + col]);
        smem[hbase + nxto + m0 * LSTR + col] = f2bf(nn0 + z0 * (ho0 - nn0));
        float r1  = fast_sigmoid(accR1[i] + pr);
        float z1  = fast_sigmoid(accZ1[i] + pz);
        float nn1 = fast_tanh(axn1[i] + pn + r1 * (ahn1[i] + bn));
        float ho1 = bf2f(smem[hbase + curo + m1 * LSTR + col]);
        smem[hbase + nxto + m1 * LSTR + col] = f2bf(nn1 + z1 * (ho1 - nn1));
      }
    }

    __syncthreads();   // h_new complete for the pair; old buffer now dead

    // ---- projection: logits[f] = h_new @ proj_w[f]^T + proj_b[f]; wave takes 4 of 8 vt ----
    for (int vv = 0; vv < NT_V / 2; ++vv) {
      const int vtu = subu * (NT_V / 2) + vv;                // SGPR-uniform
      const unsigned short* pP0 = projP + (size_t)(f * NT_V + vtu) * JT_EL + (size_t)lane * 16;
      f32x8 acc0 = {}, acc1 = {};
      for (int kb = 0; kb < KBLK; ++kb) {
        const int ho0 = hbase + nxto + mrow * LSTR + kb * 32 + khalf;
        const int ho1 = ho0 + 16 * LSTR;
        union { uint4 q[2]; bf16x16 v; } a0, a1;
        a0.q[0] = *(const uint4*)(&smem[ho0]);
        a0.q[1] = *(const uint4*)(&smem[ho0 + 16]);
        a1.q[0] = *(const uint4*)(&smem[ho1]);
        a1.q[1] = *(const uint4*)(&smem[ho1 + 16]);
        bf16x16 b = ldFrag(pP0 + (size_t)kb * FRAG_EL);
        acc0 = WMMA_BF16(a0.v, b, acc0);
        acc1 = WMMA_BF16(a1.v, b, acc1);
      }
      const int vcol = vtu * 16 + mrow;
      const float pb = proj_b[f * VV + vcol];
      #pragma unroll
      for (int i = 0; i < 8; ++i) {
        int m0 = i + mhi;
        out[((size_t)f * NROWS + (size_t)(rbase + m0)) * VV + vcol] = acc0[i] + pb;
        out[((size_t)f * NROWS + (size_t)(rbase + m0 + 16)) * VV + vcol] = acc1[i] + pb;
      }
    }

    curo = nxto;
  }
}

extern "C" void kernel_launch(void* const* d_in, const int* in_sizes, int n_in,
                              void* d_out, int out_size, void* d_ws, size_t ws_size,
                              hipStream_t stream) {
  const float* hidden = (const float*)d_in[0];
  const int*   target = (const int*)  d_in[1];
  const float* emb    = (const float*)d_in[2];
  const float* pos    = (const float*)d_in[3];
  const float* w_ih   = (const float*)d_in[4];
  const float* w_hh   = (const float*)d_in[5];
  const float* b_ih   = (const float*)d_in[6];
  const float* b_hh   = (const float*)d_in[7];
  const float* proj_w = (const float*)d_in[8];
  const float* proj_b = (const float*)d_in[9];
  float* out = (float*)d_out;

  // workspace layout
  unsigned short* wP    = (unsigned short*)d_ws;          // [w_ih | w_hh] = 2*786432 bf16 el
  unsigned short* projP = wP + 2 * WHH_EL;                // 4*8*16*512 bf16 el
  float*          posG  = (float*)(projP + 262144);       // 4*1536 f32

  pack_frags<<<3584, 32, 0, stream>>>(w_ih, w_hh, proj_w, wP, projP);
  pack_posg<<<(FF * G3 + 255) / 256, 256, 0, stream>>>(pos, w_ih, b_ih, posG);

  // 16384 rows / 32 per pair / 4 pairs per block = 128 blocks of 256 threads
  gru_main<<<NROWS / 32 / PAIRS, WAVES * 32, LDS_BYTES, stream>>>(
      hidden, target, emb, b_hh, proj_b, wP, projP, posG, out);
}
